// ScaledDotProductAttention_55336358642101
// MI455X (gfx1250) — compile-verified, matrix operand order checked by reading, not run on hardware
//
#include <hip/hip_runtime.h>

typedef __attribute__((ext_vector_type(16))) _Float16 v16h;
typedef __attribute__((ext_vector_type(8)))  _Float16 v8h;
typedef __attribute__((ext_vector_type(4)))  _Float16 v4h;
typedef __attribute__((ext_vector_type(8)))  float    v8f;
typedef __attribute__((ext_vector_type(4)))  float    v4f;

#define KB   16      // batch
#define KS   2048    // sequence length
#define KD   128     // d_model
#define KQT  128     // q rows per block (8 waves x 16)
#define KKT  64      // keys per streamed tile
#define QSTR 136     // LDS row stride (halfs) for Q (128 + 8 pad -> bank-spread)
#define KSTR 136     // LDS row stride for K tile
#define VSTR 72      // LDS row stride for V^T (64 + 8 pad)
#define PSTR 72      // LDS row stride for per-wave P strip
#define NTHREADS 256
#define NWAVES 8

// Fragment load for 16-bit A/B (16x32, K=32): per the CDNA5 VGPR layout,
// lane L (0-15) holds K = {0..7, 16..23}, lane L+16 holds K = {8..15, 24..31}
// => two contiguous 16-byte LDS reads per lane.
__device__ __forceinline__ v16h load_frag(const _Float16* rowp, int off) {
    union { v16h v; v8h h[2]; } u;
    u.h[0] = *(const v8h*)(rowp + off);
    u.h[1] = *(const v8h*)(rowp + off + 16);
    return u.v;
}

__device__ __forceinline__ v8f zero8() {
    v8f z;
    #pragma unroll
    for (int i = 0; i < 8; ++i) z[i] = 0.0f;
    return z;
}

__global__ __launch_bounds__(NTHREADS)
void sdpa_fused_wmma(const float* __restrict__ qg, const float* __restrict__ kg,
                     const float* __restrict__ vg, float* __restrict__ outg,
                     float* __restrict__ attng) {
    extern __shared__ _Float16 smem[];
    _Float16* qs = smem;                      // KQT x QSTR
    _Float16* ks = qs + KQT * QSTR;           // KKT x KSTR
    _Float16* vs = ks + KKT * KSTR;           // KD  x VSTR (V transposed: [dim][key])
    _Float16* pall = vs + KD * VSTR;          // NWAVES x 16 x PSTR

    const int tid  = threadIdx.x;
    const int lane = tid & 31;
    const int wave = tid >> 5;
    const int hi   = lane >> 4;               // half-wave select
    const int ln   = lane & 15;               // col / row-in-subtile
    const int b    = blockIdx.y;
    const int q0   = blockIdx.x * KQT;

    const float scale = 0.08838834764831845f; // 1/sqrt(128)

    // ---- Load Q tile once: f32 -> scaled f16 in LDS ----
    {
        const float* qsrc = qg + ((size_t)b * KS + q0) * KD;
        #pragma unroll
        for (int it = 0; it < (KQT * KD / 4) / NTHREADS; ++it) {
            int fi  = tid + it * NTHREADS;    // float4 index
            int row = fi >> 5;                // 32 float4 per row
            int c4  = fi & 31;
            v4f x = *(const v4f*)(qsrc + row * KD + c4 * 4);
            v4h h;
            h.x = (_Float16)(x.x * scale); h.y = (_Float16)(x.y * scale);
            h.z = (_Float16)(x.z * scale); h.w = (_Float16)(x.w * scale);
            *(v4h*)(qs + row * QSTR + c4 * 4) = h;
        }
    }
    __syncthreads();

    // ---- Hoist the wave's 4 Q A-fragments (K=32 chunks over d=128) ----
    const _Float16* qrow = qs + (wave * 16 + ln) * QSTR;
    v16h qa[4];
    #pragma unroll
    for (int kc = 0; kc < 4; ++kc) qa[kc] = load_frag(qrow, kc * 32 + hi * 8);

    float m[8], sacc[8];
    #pragma unroll
    for (int i = 0; i < 8; ++i) { m[i] = -1e30f; sacc[i] = 0.0f; }

    // ================= PASS 1: row max + per-lane partial exp-sums ===========
    for (int kt = 0; kt < KS / KKT; ++kt) {
        __syncthreads();
        const float* ksrc = kg + ((size_t)b * KS + kt * KKT) * KD;
        #pragma unroll
        for (int it = 0; it < (KKT * KD / 4) / NTHREADS; ++it) {
            int fi  = tid + it * NTHREADS;
            int row = fi >> 5;
            int c4  = fi & 31;
            v4f x = *(const v4f*)(ksrc + row * KD + c4 * 4);
            v4h h;
            h.x = (_Float16)x.x; h.y = (_Float16)x.y;
            h.z = (_Float16)x.z; h.w = (_Float16)x.w;
            *(v4h*)(ks + row * KSTR + c4 * 4) = h;
        }
        if (kt + 1 < KS / KKT)
            __builtin_prefetch(kg + ((size_t)b * KS + (kt + 1) * KKT) * KD + tid * 32, 0, 0);
        __syncthreads();

        // 4 independent accumulator chains (one per 16-key subtile), kc outer
        // so each A operand feeds 4 back-to-back WMMAs.
        v8f c[4];
        #pragma unroll
        for (int s = 0; s < 4; ++s) c[s] = zero8();
        #pragma unroll
        for (int kc = 0; kc < 4; ++kc) {
            v16h a = qa[kc];
            #pragma unroll
            for (int s = 0; s < 4; ++s) {
                v16h bb = load_frag(ks + (s * 16 + ln) * KSTR, kc * 32 + hi * 8);
                c[s] = __builtin_amdgcn_wmma_f32_16x16x32_f16(
                           false, a, false, bb, (short)0, c[s], false, false);
            }
        }

        // One 64-key max reduction per row, per-lane exp accumulation.
        #pragma unroll
        for (int i = 0; i < 8; ++i) {
            float x = fmaxf(fmaxf(c[0][i], c[1][i]), fmaxf(c[2][i], c[3][i]));
            #pragma unroll
            for (int off = 1; off < 16; off <<= 1)
                x = fmaxf(x, __shfl_xor(x, off, 16));
            float mnew = fmaxf(m[i], x);
            float r = __expf(m[i] - mnew);
            float e = __expf(c[0][i] - mnew) + __expf(c[1][i] - mnew)
                    + __expf(c[2][i] - mnew) + __expf(c[3][i] - mnew);
            sacc[i] = sacc[i] * r + e;
            m[i] = mnew;
        }
    }

    // Single cross-lane sum reduction for the softmax denominators.
    float invl[8];
    #pragma unroll
    for (int i = 0; i < 8; ++i) {
        float s = sacc[i];
        #pragma unroll
        for (int off = 1; off < 16; off <<= 1)
            s += __shfl_xor(s, off, 16);
        invl[i] = 1.0f / s;
    }

    v8f oacc[8];
    #pragma unroll
    for (int n = 0; n < 8; ++n) oacc[n] = zero8();

    _Float16* pw = pall + wave * 16 * PSTR;

    // ============ PASS 2: normalized attn write + out = P @ V ============
    for (int kt = 0; kt < KS / KKT; ++kt) {
        __syncthreads();
        const float* ksrc = kg + ((size_t)b * KS + kt * KKT) * KD;
        const float* vsrc = vg + ((size_t)b * KS + kt * KKT) * KD;
        #pragma unroll
        for (int it = 0; it < (KKT * KD / 4) / NTHREADS; ++it) {
            int fi  = tid + it * NTHREADS;
            int row = fi >> 5;
            int c4  = fi & 31;
            v4f x = *(const v4f*)(ksrc + row * KD + c4 * 4);
            v4h h;
            h.x = (_Float16)x.x; h.y = (_Float16)x.y;
            h.z = (_Float16)x.z; h.w = (_Float16)x.w;
            *(v4h*)(ks + row * KSTR + c4 * 4) = h;
            v4f y = *(const v4f*)(vsrc + row * KD + c4 * 4);
            vs[(c4 * 4 + 0) * VSTR + row] = (_Float16)y.x;   // transpose into [dim][key]
            vs[(c4 * 4 + 1) * VSTR + row] = (_Float16)y.y;
            vs[(c4 * 4 + 2) * VSTR + row] = (_Float16)y.z;
            vs[(c4 * 4 + 3) * VSTR + row] = (_Float16)y.w;
        }
        if (kt + 1 < KS / KKT) {
            __builtin_prefetch(kg + ((size_t)b * KS + (kt + 1) * KKT) * KD + tid * 32, 0, 0);
            __builtin_prefetch(vg + ((size_t)b * KS + (kt + 1) * KKT) * KD + tid * 32, 0, 0);
        }
        __syncthreads();

        v8f c[4];
        #pragma unroll
        for (int s = 0; s < 4; ++s) c[s] = zero8();
        #pragma unroll
        for (int kc = 0; kc < 4; ++kc) {
            v16h a = qa[kc];
            #pragma unroll
            for (int s = 0; s < 4; ++s) {
                v16h bb = load_frag(ks + (s * 16 + ln) * KSTR, kc * 32 + hi * 8);
                c[s] = __builtin_amdgcn_wmma_f32_16x16x32_f16(
                           false, a, false, bb, (short)0, c[s], false, false);
            }
        }

        #pragma unroll
        for (int s = 0; s < 4; ++s) {
            #pragma unroll
            for (int i = 0; i < 8; ++i) {
                float p = __expf(c[s][i] - m[i]) * invl[i];
                size_t qr = (size_t)(q0 + wave * 16 + i + 8 * hi);
                attng[((size_t)b * KS + qr) * KS + (size_t)kt * KKT + s * 16 + ln] = p;
                pw[(i + 8 * hi) * PSTR + s * 16 + ln] = (_Float16)p;
            }
        }

        // Second GEMM over this key tile: out += P(16x64) @ V(64x128).
        // Same-wave DS ops are in-order, so the p-strip reads below see the
        // writes above without an extra barrier.
        #pragma unroll
        for (int kc2 = 0; kc2 < 2; ++kc2) {     // two K=32 chunks over 64 keys
            v16h a = load_frag(pw + ln * PSTR, kc2 * 32 + hi * 8);
            #pragma unroll
            for (int n = 0; n < 8; ++n) {
                v16h bb = load_frag(vs + (n * 16 + ln) * VSTR, kc2 * 32 + hi * 8);
                oacc[n] = __builtin_amdgcn_wmma_f32_16x16x32_f16(
                              false, a, false, bb, (short)0, oacc[n], false, false);
            }
        }
    }

    // ---- write out [B, S, D] ----
    #pragma unroll
    for (int n = 0; n < 8; ++n) {
        #pragma unroll
        for (int i = 0; i < 8; ++i) {
            size_t qr = (size_t)(q0 + wave * 16 + i + 8 * hi);
            outg[((size_t)b * KS + qr) * KD + n * 16 + ln] = oacc[n][i];
        }
    }
}

extern "C" void kernel_launch(void* const* d_in, const int* in_sizes, int n_in,
                              void* d_out, int out_size, void* d_ws, size_t ws_size,
                              hipStream_t stream) {
    const float* q = (const float*)d_in[0];
    const float* k = (const float*)d_in[1];
    const float* v = (const float*)d_in[2];
    float* out  = (float*)d_out;                       // [B, S, D]
    float* attn = out + (size_t)KB * KS * KD;          // [B, S, S]

    dim3 grid(KS / KQT, KB);
    size_t shmem = (size_t)(KQT * QSTR + KKT * KSTR + KD * VSTR + NWAVES * 16 * PSTR)
                 * sizeof(_Float16);                   // ~87 KB, fits 320 KB WGP LDS
    hipLaunchKernelGGL(sdpa_fused_wmma, grid, dim3(NTHREADS), shmem, stream,
                       q, k, v, out, attn);
}
